// Nerfies_Deformation_63144609186213
// MI455X (gfx1250) — matrix operand order repaired and checked by reading, not compiled, not run on hardware
//
#include <hip/hip_runtime.h>

typedef __attribute__((ext_vector_type(16))) _Float16 v16h;
typedef __attribute__((ext_vector_type(8)))  _Float16 v8h;
typedef __attribute__((ext_vector_type(8)))  float    v8f;

#define WIDTH   128
#define NPB     32          // points per block
#define ROWS    128         // 32 points * (value + 3 tangents)
#define STRIDE  136         // halves; 272B row stride keeps b128 LDS loads 16B aligned
#define PI_LIT  3.14f       // reference uses literal 3.14
#define WIN_PAD 8192        // 128 x 64 padded f16 input weights
#define WHID    16384       // 128 x 128 f16 per hidden layer

union HFrag { v16h v; struct { v8h lo, hi; } p; };

// ---------------- dual numbers: value + d/dx,d/dy,d/dz ----------------
struct D4 { float v, dx, dy, dz; };
__device__ __forceinline__ D4 mk(float a)            { return {a, 0.f, 0.f, 0.f}; }
__device__ __forceinline__ D4 dadd(D4 a, D4 b)       { return {a.v+b.v, a.dx+b.dx, a.dy+b.dy, a.dz+b.dz}; }
__device__ __forceinline__ D4 dsub(D4 a, D4 b)       { return {a.v-b.v, a.dx-b.dx, a.dy-b.dy, a.dz-b.dz}; }
__device__ __forceinline__ D4 dmul(D4 a, D4 b)       { return {a.v*b.v, a.v*b.dx+a.dx*b.v, a.v*b.dy+a.dy*b.v, a.v*b.dz+a.dz*b.v}; }
__device__ __forceinline__ D4 dneg(D4 a)             { return {-a.v, -a.dx, -a.dy, -a.dz}; }
__device__ __forceinline__ D4 ddiv(D4 a, D4 b) {
  float iv = 1.f / b.v; float q = a.v * iv;
  return {q, (a.dx - q*b.dx)*iv, (a.dy - q*b.dy)*iv, (a.dz - q*b.dz)*iv};
}
__device__ __forceinline__ D4 dsqrt_(D4 a) {
  float r = sqrtf(a.v); float h = 0.5f / r;
  return {r, a.dx*h, a.dy*h, a.dz*h};
}
__device__ __forceinline__ D4 dsin_(D4 a) {
  float s = __sinf(a.v), c = __cosf(a.v);
  return {s, c*a.dx, c*a.dy, c*a.dz};
}
__device__ __forceinline__ D4 dcos_(D4 a) {
  float s = __sinf(a.v), c = __cosf(a.v);
  return {c, -s*a.dx, -s*a.dy, -s*a.dz};
}

// ---------------- weight conversion pre-pass (f32 -> f16 pack in d_ws) ---
__global__ void convert_weights(const float* __restrict__ Win,
                                const float* __restrict__ Ws,
                                _Float16* __restrict__ out) {
  int i = blockIdx.x * 256 + threadIdx.x;
  if (i < WIN_PAD) {                         // [128][64] zero-padded from [128][39]
    int o = i >> 6, k = i & 63;
    out[i] = (_Float16)(k < 39 ? Win[o*39 + k] : 0.f);
  } else if (i < WIN_PAD + 6*WHID) {         // [6][128][128]
    out[i] = (_Float16)Ws[i - WIN_PAD];
  }
}

// ---------------- one fused MLP layer on WMMA ----------------
// KT = K/32 (2 for input layer K=64, 4 for hidden K=128). Output width fixed 128.
// N-tiles are processed in pairs as two independent accumulator chains so the
// scheduler can fill WMMA hazard/co-exec slots with the partner chain instead
// of v_nops.
template<int KT>
__device__ __forceinline__ void mlp_layer(_Float16* sAct,
                                          const _Float16* __restrict__ Wl,
                                          const float* __restrict__ bias,
                                          float slope, int tid) {
  const int wave = tid >> 5, lane = tid & 31;
  const int lrow = lane & 15;
  const int hi   = (lane & 16) ? 1 : 0;

  // ---- load A fragments (own 16-row M-tile) into registers BEFORE barrier
  HFrag A[KT];
  {
    const _Float16* rp = &sAct[(wave*16 + lrow) * STRIDE];
    const int boff = hi ? 8 : 0;
#pragma unroll
    for (int kt = 0; kt < KT; ++kt) {
      A[kt].p.lo = *(const v8h*)(rp + kt*32 + boff);
      A[kt].p.hi = *(const v8h*)(rp + kt*32 + boff + 16);
    }
  }
  __syncthreads();   // everyone has A in registers -> safe to overwrite sAct

  // branch-free bias: always load, scale by wave-uniform 0/1
  const float biasScale = (wave < 2) ? 1.f : 0.f;   // M-tiles 0,1 = value rows
  const int  K = KT * 32;
  const int  mbase = wave*16 + (hi ? 8 : 0);
  const int  kb = hi ? 8 : 0;

#pragma unroll
  for (int np = 0; np < 4; ++np) {
    const int nt0 = np*2, nt1 = np*2 + 1;
    const int o0 = nt0*16 + lrow;
    const int o1 = nt1*16 + lrow;
    const float b0 = bias[o0] * biasScale;
    const float b1 = bias[o1] * biasScale;
    v8f c0, c1;
#pragma unroll
    for (int r = 0; r < 8; ++r) { c0[r] = b0; c1[r] = b1; }
#pragma unroll
    for (int kt = 0; kt < KT; ++kt) {
      HFrag B0, B1;
      const v8h* bp0 = (const v8h*)(Wl + o0*K + kt*32 + kb);
      const v8h* bp1 = (const v8h*)(Wl + o1*K + kt*32 + kb);
      B0.p.lo = bp0[0]; B0.p.hi = bp0[2];
      B1.p.lo = bp1[0]; B1.p.hi = bp1[2];
      c0 = __builtin_amdgcn_wmma_f32_16x16x32_f16(false, A[kt].v, false, B0.v,
                                                  (short)0, c0, false, false);
      c1 = __builtin_amdgcn_wmma_f32_16x16x32_f16(false, A[kt].v, false, B1.v,
                                                  (short)0, c1, false, false);
    }
    // D fragments -> LDS (f16 pre-activation)
#pragma unroll
    for (int r = 0; r < 8; ++r) {
      sAct[(mbase + r)*STRIDE + nt0*16 + lrow] = (_Float16)c0[r];
      sAct[(mbase + r)*STRIDE + nt1*16 + lrow] = (_Float16)c1[r];
    }
  }
  __syncthreads();

  // ---- activation in place. value rows: (leaky)relu; tangent rows: * act'(z_value)
  // Sign of the value-row entry is invariant under (leaky)relu, so the mask read
  // is race-free regardless of write ordering.
  {
    int row = tid >> 1;
    int c0  = (tid & 1) * 64;
    bool isVal = row < NPB;
    int vrow = row & 31;
    for (int j = 0; j < 64; ++j) {
      int col = c0 + j;
      float z = (float)sAct[row*STRIDE + col];
      float oV;
      if (isVal) {
        oV = z > 0.f ? z : slope * z;
      } else {
        float zv = (float)sAct[vrow*STRIDE + col];
        oV = z * (zv > 0.f ? 1.f : slope);
      }
      sAct[row*STRIDE + col] = (_Float16)oV;
    }
  }
  __syncthreads();
}

// ---------------- fused deformation kernel ----------------
__global__ __launch_bounds__(256)
void nerfies_fused(const float* __restrict__ X,
                   const _Float16* __restrict__ Wpack,
                   const float* __restrict__ b_in,
                   const float* __restrict__ bs,
                   const float* __restrict__ Ww, const float* __restrict__ bw,
                   const float* __restrict__ Wv, const float* __restrict__ bv,
                   const int* __restrict__ iterPtr,
                   float* __restrict__ out, int N) {
  __shared__ _Float16 sAct[ROWS * STRIDE];
  __shared__ float    sHead[ROWS * 6];

  const int tid = threadIdx.x;
  const int blockPt = blockIdx.x * NPB;

  // ================= positional encoding (value + 3 tangent rows) ========
  if (tid < ROWS) {
    int row = tid;
    int p = row & 31;
    int n = blockPt + p;
    float px[3] = {X[3*n], X[3*n+1], X[3*n+2]};
    float it   = (float)(*iterPtr);
    float aAnn = 6.0f * it / 3000.0f;              // M*it / (0.6*5000)

    // per-frequency anneal weight and scale, computed once
    float wa[6], ml[6];
#pragma unroll
    for (int j = 0; j < 6; ++j) {
      float t = fminf(fmaxf(aAnn - (float)j, 0.f), 1.f);
      wa[j] = (1.f - __cosf(t * PI_LIT)) * 0.5f;
      ml[j] = PI_LIT * exp2f((float)(j - 3));
    }

    _Float16* rp = &sAct[row * STRIDE];
    if (row < NPB) {                               // value row
      rp[0] = (_Float16)px[0]; rp[1] = (_Float16)px[1]; rp[2] = (_Float16)px[2];
      int c = 3;
      for (int ax = 0; ax < 3; ++ax) {
#pragma unroll
        for (int j = 0; j < 6; ++j) {
          float ang = px[ax] * ml[j];
          rp[c + j]     = (_Float16)(__sinf(ang) * wa[j]);
          rp[c + 6 + j] = (_Float16)(__cosf(ang) * wa[j]);
        }
        c += 12;
      }
      for (int k = 39; k < 64; ++k) rp[k] = (_Float16)0.f;
    } else {                                       // tangent row, direction d
      int d = (row - 32) >> 5;
      rp[0] = (_Float16)(d == 0 ? 1.f : 0.f);
      rp[1] = (_Float16)(d == 1 ? 1.f : 0.f);
      rp[2] = (_Float16)(d == 2 ? 1.f : 0.f);
      int c = 3;
      for (int ax = 0; ax < 3; ++ax) {
#pragma unroll
        for (int j = 0; j < 6; ++j) {
          float vs = 0.f, vc = 0.f;
          if (ax == d) {
            float ang = px[ax] * ml[j];
            vs =  __cosf(ang) * ml[j] * wa[j];     // d/dx sin(x*ml)*wa
            vc = -__sinf(ang) * ml[j] * wa[j];     // d/dx cos(x*ml)*wa
          }
          rp[c + j]     = (_Float16)vs;
          rp[c + 6 + j] = (_Float16)vc;
        }
        c += 12;
      }
      for (int k = 39; k < 64; ++k) rp[k] = (_Float16)0.f;
    }
  }
  __syncthreads();

  // ================= MLP: input layer (K=64, leaky 0.01) + 6 hidden =======
  __builtin_prefetch(Wpack + WIN_PAD, 0, 1);
  mlp_layer<2>(sAct, Wpack, b_in, 0.01f, tid);
  for (int l = 0; l < 6; ++l) {
    if (l < 5) __builtin_prefetch(Wpack + WIN_PAD + (l + 1) * WHID, 0, 1);
    mlp_layer<4>(sAct, Wpack + WIN_PAD + l * WHID, bs + l * WIDTH, 0.f, tid);
  }

  // ================= heads: w,v = fea @ {Ww,Wv}^T (+bias on value rows) ====
  if (tid < ROWS) {
    int row = tid;
    float bScale = row < NPB ? 1.f : 0.f;
    float acc[6];
#pragma unroll
    for (int o = 0; o < 3; ++o) {
      acc[o]     = bw[o] * bScale;
      acc[3 + o] = bv[o] * bScale;
    }
    for (int i = 0; i < WIDTH; ++i) {
      float f = (float)sAct[row * STRIDE + i];
#pragma unroll
      for (int o = 0; o < 3; ++o) {
        acc[o]     += f * Ww[o * WIDTH + i];
        acc[3 + o] += f * Wv[o * WIDTH + i];
      }
    }
#pragma unroll
    for (int o = 0; o < 6; ++o) sHead[row * 6 + o] = acc[o];
  }
  __syncthreads();

  // ================= SE(3) exp + warp + Jacobian (dual numbers) ===========
  if (tid < NPB) {
    int p = tid, n = blockPt + p;
    D4 w[3], vv[3];
#pragma unroll
    for (int o = 0; o < 3; ++o) {
      w[o]  = { sHead[p*6 + o],     sHead[(32+p)*6 + o],
                sHead[(64+p)*6 + o], sHead[(96+p)*6 + o] };
      vv[o] = { sHead[p*6 + 3 + o],     sHead[(32+p)*6 + 3 + o],
                sHead[(64+p)*6 + 3 + o], sHead[(96+p)*6 + 3 + o] };
    }
    D4 th2 = dadd(dadd(dmul(w[0], w[0]), dmul(w[1], w[1])), dmul(w[2], w[2]));
    D4 th  = dsqrt_(th2);
    D4 wn[3], vn[3];
#pragma unroll
    for (int o = 0; o < 3; ++o) { wn[o] = ddiv(w[o], th); vn[o] = ddiv(vv[o], th); }

    D4 s   = dsin_(th);
    D4 cth = dcos_(th);
    D4 omc = dsub(mk(1.f), cth);
    D4 tms = dsub(th, s);
    D4 Z   = mk(0.f);

    D4 Wm[3][3] = {
      { Z,            dneg(wn[2]), wn[1]       },
      { wn[2],        Z,           dneg(wn[0]) },
      { dneg(wn[1]),  wn[0],       Z           } };
    D4 W2[3][3];
#pragma unroll
    for (int i = 0; i < 3; ++i)
#pragma unroll
      for (int j = 0; j < 3; ++j)
        W2[i][j] = dadd(dadd(dmul(Wm[i][0], Wm[0][j]),
                             dmul(Wm[i][1], Wm[1][j])),
                        dmul(Wm[i][2], Wm[2][j]));

    D4 xd[3];
#pragma unroll
    for (int j = 0; j < 3; ++j)
      xd[j] = { X[3*n + j], j == 0 ? 1.f : 0.f, j == 1 ? 1.f : 0.f, j == 2 ? 1.f : 0.f };

    float* oW = out;
    float* oJ = out + (size_t)N * 3;
#pragma unroll
    for (int i = 0; i < 3; ++i) {
      D4 Ri[3], Gi[3];
#pragma unroll
      for (int j = 0; j < 3; ++j) {
        Ri[j] = dadd((i == j ? mk(1.f) : Z),
                     dadd(dmul(s, Wm[i][j]), dmul(omc, W2[i][j])));
        Gi[j] = dadd((i == j ? th : Z),
                     dadd(dmul(omc, Wm[i][j]), dmul(tms, W2[i][j])));
      }
      D4 ti  = dadd(dadd(dmul(Gi[0], vn[0]), dmul(Gi[1], vn[1])), dmul(Gi[2], vn[2]));
      D4 res = dadd(dadd(dadd(dmul(Ri[0], xd[0]), dmul(Ri[1], xd[1])),
                         dmul(Ri[2], xd[2])), ti);
      oW[(size_t)n*3 + i]          = res.v;
      oJ[(size_t)n*9 + i*3 + 0]    = res.dx;
      oJ[(size_t)n*9 + i*3 + 1]    = res.dy;
      oJ[(size_t)n*9 + i*3 + 2]    = res.dz;
    }
  }
}

extern "C" void kernel_launch(void* const* d_in, const int* in_sizes, int n_in,
                              void* d_out, int out_size, void* d_ws, size_t ws_size,
                              hipStream_t stream) {
  (void)n_in; (void)out_size; (void)ws_size;
  const float* X   = (const float*)d_in[0];
  const float* Win = (const float*)d_in[1];
  const float* bin = (const float*)d_in[2];
  const float* Ws  = (const float*)d_in[3];
  const float* bsP = (const float*)d_in[4];
  const float* Ww  = (const float*)d_in[5];
  const float* bw  = (const float*)d_in[6];
  const float* Wv  = (const float*)d_in[7];
  const float* bv  = (const float*)d_in[8];
  const int*   itp = (const int*)d_in[9];

  int N = in_sizes[0] / 3;                 // 131072
  _Float16* wpack = (_Float16*)d_ws;       // [WIN_PAD + 6*WHID] halves (~213 KB)

  int convTotal = WIN_PAD + 6 * WHID;
  convert_weights<<<(convTotal + 255) / 256, 256, 0, stream>>>(Win, Ws, wpack);
  nerfies_fused<<<N / NPB, 256, 0, stream>>>(X, wpack, bin, bsP, Ww, bw, Wv, bv,
                                             itp, (float*)d_out, N);
}
